// MaskedCrossAttention_25434796327656
// MI455X (gfx1250) — matrix operand order repaired and checked by reading, not compiled, not run on hardware
//
#include <hip/hip_runtime.h>
#include <hip/hip_bf16.h>
#include <math.h>

// ---------------------------------------------------------------------------
// MaskedCrossAttention for MI455X (gfx1250, wave32, WMMA)
//   LN -> GEMM(q) -> GEMM(kv) -> block-diag masked attention -> GEMM(out)
// GEMMs use v_wmma_f32_16x16x32_f16 (f32 accumulate, f16 operands).
// ---------------------------------------------------------------------------

typedef __attribute__((ext_vector_type(16))) _Float16 v16h;
typedef __attribute__((ext_vector_type(8)))  _Float16 v8h;
typedef __attribute__((ext_vector_type(8)))  float    v8f;

#define B_     4
#define TQ_    2048
#define DIM_   1024
#define T_IMG_ 16
#define N_LAT_ 64
#define D_LAT_ 1024
#define HEADS_ 8
#define DHEAD_ 64
#define INNER_ 512          // HEADS_*DHEAD_
#define NKV_   1024         // T_IMG_*N_LAT_

// ---------------------------------------------------------------------------
// LayerNorm: one block (256 threads) per row; output f16
// ---------------------------------------------------------------------------
__global__ void ln_kernel(const float* __restrict__ x,
                          const float* __restrict__ gamma,
                          const float* __restrict__ beta,
                          _Float16* __restrict__ xn)
{
    const int row = blockIdx.x;
    const float* xr = x + (size_t)row * DIM_;
    __shared__ float rs[256], rs2[256];
    float s = 0.f, s2 = 0.f;
    for (int i = threadIdx.x; i < DIM_; i += 256) {
        float v = xr[i]; s += v; s2 += v * v;
    }
    rs[threadIdx.x] = s; rs2[threadIdx.x] = s2;
    __syncthreads();
    for (int off = 128; off > 0; off >>= 1) {
        if ((int)threadIdx.x < off) {
            rs[threadIdx.x]  += rs[threadIdx.x + off];
            rs2[threadIdx.x] += rs2[threadIdx.x + off];
        }
        __syncthreads();
    }
    const float mu  = rs[0]  * (1.f / DIM_);
    const float var = rs2[0] * (1.f / DIM_) - mu * mu;
    const float inv = rsqrtf(var + 1e-5f);
    _Float16* xo = xn + (size_t)row * DIM_;
    for (int i = threadIdx.x; i < DIM_; i += 256)
        xo[i] = (_Float16)((xr[i] - mu) * inv * gamma[i] + beta[i]);
}

// ---------------------------------------------------------------------------
// Convert f32 [K][N] -> f16 transposed [N][K]  (so GEMM B-operand is K-contig)
// ---------------------------------------------------------------------------
__global__ void tconv_kernel(const float* __restrict__ in, _Float16* __restrict__ out,
                             int K, int N)
{
    int i = blockIdx.x * 256 + threadIdx.x;
    if (i >= K * N) return;
    int k = i / N, n = i - k * N;
    out[(size_t)n * K + k] = (_Float16)in[i];
}

__global__ void conv_kernel(const float* __restrict__ in, _Float16* __restrict__ out, int n)
{
    int i = blockIdx.x * 256 + threadIdx.x;
    if (i < n) out[i] = (_Float16)in[i];
}

// ---------------------------------------------------------------------------
// q_time = cumsum(media_locations) along T, per batch (tiny)
// ---------------------------------------------------------------------------
__global__ void qtime_kernel(const int* __restrict__ ml, int* __restrict__ qt)
{
    const int b = blockIdx.x;
    if (threadIdx.x == 0) {
        int c = 0;
        for (int t = 0; t < TQ_; ++t) {
            c += (ml[b * TQ_ + t] != 0) ? 1 : 0;
            qt[b * TQ_ + t] = c;
        }
    }
}

// ---------------------------------------------------------------------------
// WMMA GEMM: C[M][N] = A[M][K] * BT[N][K]^T
//   wave -> 16x64 C tile (4 accumulators, shared A fragment)
//   block = 8 waves -> 128x64 C tile.  M%128==0, N%64==0, K%32==0.
// fragment layouts used here (wave32; contiguous k, A and B alike):
//   A 16x32 f16 : lane m=lane&15; elems 0..7 -> K=k0+16*hi.. ; 8..15 -> +8
//   B 32x16 f16 : lane n=lane&15; elems 0..15 -> K=k0+16*hi..
//   C 16x16 f32 : elem r -> M=r+8*hi, N=lane&15
// ---------------------------------------------------------------------------
template <typename OutT>
__global__ void wmma_gemm_kernel(const _Float16* __restrict__ A,
                                 const _Float16* __restrict__ BT,
                                 OutT* __restrict__ C,
                                 int M, int N, int K)
{
    const int wave = threadIdx.x >> 5;
    const int lane = threadIdx.x & 31;
    const int hi   = lane >> 4;
    const int l16  = lane & 15;
    const int row0 = (blockIdx.y * 8 + wave) * 16;
    const int col0 = blockIdx.x * 64;
    if (row0 >= M || col0 >= N) return;

    v8f acc0 = {}, acc1 = {}, acc2 = {}, acc3 = {};
    const _Float16* arow = A + (size_t)(row0 + l16) * K;
    const _Float16* b0 = BT + (size_t)(col0 +  0 + l16) * K + 16 * hi;
    const _Float16* b1 = BT + (size_t)(col0 + 16 + l16) * K + 16 * hi;
    const _Float16* b2 = BT + (size_t)(col0 + 32 + l16) * K + 16 * hi;
    const _Float16* b3 = BT + (size_t)(col0 + 48 + l16) * K + 16 * hi;

    for (int k0 = 0; k0 < K; k0 += 32) {
        v8h alo = *(const v8h*)(arow + k0 + 16 * hi);
        v8h ahi = *(const v8h*)(arow + k0 + 16 * hi + 8);
        v16h a;
#pragma unroll
        for (int e = 0; e < 8; ++e) { a[e] = alo[e]; a[e + 8] = ahi[e]; }
        v16h vb0 = *(const v16h*)(b0 + k0);
        v16h vb1 = *(const v16h*)(b1 + k0);
        v16h vb2 = *(const v16h*)(b2 + k0);
        v16h vb3 = *(const v16h*)(b3 + k0);
        acc0 = __builtin_amdgcn_wmma_f32_16x16x32_f16(false, a, false, vb0, (short)0, acc0, false, false);
        acc1 = __builtin_amdgcn_wmma_f32_16x16x32_f16(false, a, false, vb1, (short)0, acc1, false, false);
        acc2 = __builtin_amdgcn_wmma_f32_16x16x32_f16(false, a, false, vb2, (short)0, acc2, false, false);
        acc3 = __builtin_amdgcn_wmma_f32_16x16x32_f16(false, a, false, vb3, (short)0, acc3, false, false);
    }
#pragma unroll
    for (int r = 0; r < 8; ++r) {
        const size_t m = (size_t)(row0 + r + 8 * hi);
        OutT* crow = C + m * N + col0 + l16;
        crow[ 0] = (OutT)acc0[r];
        crow[16] = (OutT)acc1[r];
        crow[32] = (OutT)acc2[r];
        crow[48] = (OutT)acc3[r];
    }
}

// ---------------------------------------------------------------------------
// Attention: block = (b, h, 128-query block), 128 threads (1 thread / query).
// Each query attends to exactly the 64 KV rows of its image time (block-diag
// mask).  K/V tile staged in LDS (block-uniform q_time; global fallback if a
// thread diverges).  Two-pass softmax over 64 entries.
// ---------------------------------------------------------------------------
__global__ void attn_kernel(const _Float16* __restrict__ qh,   // [B*TQ][INNER]
                            const _Float16* __restrict__ kvh,  // [B*NKV][2*INNER]
                            const int* __restrict__ qt,        // [B*TQ]
                            _Float16* __restrict__ ao)         // [B*TQ][INNER]
{
    const int b = blockIdx.z, h = blockIdx.y, qblk = blockIdx.x;
    const int q   = qblk * 128 + threadIdx.x;
    const int tq  = qt[b * TQ_ + q];
    const int tq0 = qt[b * TQ_ + qblk * 128];
    const float scale = 0.125f;   // DIM_HEAD^-0.5

    __shared__ __align__(16) _Float16 ks[64 * 64];
    __shared__ __align__(16) _Float16 vs[64 * 64];
    if (tq0 > 0) {
        const int jbase = (tq0 - 1) * 64;
        for (int c = threadIdx.x; c < 512; c += 128) {
            const int j = c >> 3, o = (c & 7) * 8;
            const size_t base = ((size_t)(b * NKV_ + jbase + j)) * (2 * INNER_) + h * DHEAD_ + o;
            *(v8h*)(ks + j * 64 + o) = *(const v8h*)(kvh + base);
            *(v8h*)(vs + j * 64 + o) = *(const v8h*)(kvh + base + INNER_);
        }
    }
    __syncthreads();

    float outv[64];
#pragma unroll
    for (int d = 0; d < 64; ++d) outv[d] = 0.f;

    if (tq > 0) {
        v8h qv8[8];
        const _Float16* qr = qh + ((size_t)(b * TQ_ + q)) * INNER_ + h * DHEAD_;
#pragma unroll
        for (int d8 = 0; d8 < 8; ++d8) qv8[d8] = *(const v8h*)(qr + d8 * 8);

        const bool inl = (tq == tq0);
        const size_t gk = ((size_t)(b * NKV_ + (tq - 1) * 64)) * (2 * INNER_) + h * DHEAD_;

        // pass 1: row max
        float mx = -3.0e38f;
        for (int j = 0; j < 64; ++j) {
            const _Float16* kr = inl ? (ks + j * 64)
                                     : (kvh + gk + (size_t)j * (2 * INNER_));
            float s = 0.f;
#pragma unroll
            for (int d8 = 0; d8 < 8; ++d8) {
                v8h kk = *(const v8h*)(kr + d8 * 8);
#pragma unroll
                for (int e = 0; e < 8; ++e) s += (float)qv8[d8][e] * (float)kk[e];
            }
            mx = fmaxf(mx, s * scale);
        }
        // pass 2: exp/sum + weighted V accumulate
        float l = 0.f;
        for (int j = 0; j < 64; ++j) {
            const _Float16* kr = inl ? (ks + j * 64)
                                     : (kvh + gk + (size_t)j * (2 * INNER_));
            float s = 0.f;
#pragma unroll
            for (int d8 = 0; d8 < 8; ++d8) {
                v8h kk = *(const v8h*)(kr + d8 * 8);
#pragma unroll
                for (int e = 0; e < 8; ++e) s += (float)qv8[d8][e] * (float)kk[e];
            }
            const float p = __expf(s * scale - mx);
            l += p;
            const _Float16* vr = inl ? (vs + j * 64)
                                     : (kvh + gk + INNER_ + (size_t)j * (2 * INNER_));
#pragma unroll
            for (int d8 = 0; d8 < 8; ++d8) {
                v8h vv = *(const v8h*)(vr + d8 * 8);
#pragma unroll
                for (int e = 0; e < 8; ++e) outv[d8 * 8 + e] += p * (float)vv[e];
            }
        }
        const float rl = 1.f / l;
#pragma unroll
        for (int d = 0; d < 64; ++d) outv[d] *= rl;
    }

    _Float16* aor = ao + ((size_t)(b * TQ_ + q)) * INNER_ + h * DHEAD_;
#pragma unroll
    for (int d8 = 0; d8 < 8; ++d8) {
        v8h ov;
#pragma unroll
        for (int e = 0; e < 8; ++e) ov[e] = (_Float16)outv[d8 * 8 + e];
        *(v8h*)(aor + d8 * 8) = ov;
    }
}

// ---------------------------------------------------------------------------
// Host launcher
// ---------------------------------------------------------------------------
extern "C" void kernel_launch(void* const* d_in, const int* in_sizes, int n_in,
                              void* d_out, int out_size, void* d_ws, size_t ws_size,
                              hipStream_t stream)
{
    const float* x     = (const float*)d_in[0];
    const float* media = (const float*)d_in[1];
    const int*   ml    = (const int*)d_in[2];   // media_locations (0/1)
    const float* gamma = (const float*)d_in[3];
    const float* beta  = (const float*)d_in[4];
    const float* Wq    = (const float*)d_in[5];
    const float* Wkv   = (const float*)d_in[6];
    const float* Wout  = (const float*)d_in[7];
    float* out = (float*)d_out;
    (void)in_sizes; (void)n_in; (void)out_size; (void)ws_size;

    char* ws = (char*)d_ws;
    size_t off = 0;
    auto alloc = [&](size_t bytes) -> void* {
        void* p = ws + off;
        off += (bytes + 255) & ~(size_t)255;
        return p;
    };
    _Float16* xn_h  = (_Float16*)alloc((size_t)B_ * TQ_ * DIM_ * 2);
    _Float16* WqT   = (_Float16*)alloc((size_t)INNER_ * DIM_ * 2);
    _Float16* WkvT  = (_Float16*)alloc((size_t)2 * INNER_ * D_LAT_ * 2);
    _Float16* WoutT = (_Float16*)alloc((size_t)DIM_ * INNER_ * 2);
    _Float16* med_h = (_Float16*)alloc((size_t)B_ * NKV_ * D_LAT_ * 2);
    _Float16* q_h   = (_Float16*)alloc((size_t)B_ * TQ_ * INNER_ * 2);
    _Float16* kv_h  = (_Float16*)alloc((size_t)B_ * NKV_ * 2 * INNER_ * 2);
    _Float16* ao_h  = (_Float16*)alloc((size_t)B_ * TQ_ * INNER_ * 2);
    int*      qtime = (int*)alloc((size_t)B_ * TQ_ * 4);
    // total ~55 MB of workspace

    // 1) LayerNorm -> f16
    ln_kernel<<<B_ * TQ_, 256, 0, stream>>>(x, gamma, beta, xn_h);
    // 2) weights: f32 -> f16 transposed [N][K]
    tconv_kernel<<<(DIM_ * INNER_ + 255) / 256, 256, 0, stream>>>(Wq, WqT, DIM_, INNER_);
    tconv_kernel<<<(D_LAT_ * 2 * INNER_ + 255) / 256, 256, 0, stream>>>(Wkv, WkvT, D_LAT_, 2 * INNER_);
    tconv_kernel<<<(INNER_ * DIM_ + 255) / 256, 256, 0, stream>>>(Wout, WoutT, INNER_, DIM_);
    // 3) media -> f16
    conv_kernel<<<(B_ * NKV_ * D_LAT_ + 255) / 256, 256, 0, stream>>>(media, med_h, B_ * NKV_ * D_LAT_);
    // 4) q_time prefix sums
    qtime_kernel<<<B_, 32, 0, stream>>>(ml, qtime);
    // 5) q = xn @ Wq   (8192x512, K=1024), f16 out
    wmma_gemm_kernel<_Float16><<<dim3(INNER_ / 64, (B_ * TQ_) / 128), 256, 0, stream>>>(
        xn_h, WqT, q_h, B_ * TQ_, INNER_, DIM_);
    // 6) kv = media @ Wkv  (4096x1024, K=1024), f16 out (k = cols 0..511, v = 512..1023)
    wmma_gemm_kernel<_Float16><<<dim3((2 * INNER_) / 64, (B_ * NKV_) / 128), 256, 0, stream>>>(
        med_h, WkvT, kv_h, B_ * NKV_, 2 * INNER_, D_LAT_);
    // 7) masked attention -> ao (f16)
    attn_kernel<<<dim3(TQ_ / 128, HEADS_, B_), 128, 0, stream>>>(q_h, kv_h, qtime, ao_h);
    // 8) out = ao @ Wout  (8192x1024, K=512), f32 out
    wmma_gemm_kernel<float><<<dim3(DIM_ / 64, (B_ * TQ_) / 128), 256, 0, stream>>>(
        ao_h, WoutT, out, B_ * TQ_, DIM_, INNER_);
}
